// DenseGeneral_76304388981477
// MI455X (gfx1250) — 8-bit matrix operand order repaired and checked by reading, compiled, not run on hardware
//
#include <hip/hip_runtime.h>

typedef __attribute__((ext_vector_type(8))) int   v8i;
typedef __attribute__((ext_vector_type(4))) int   v4i;
typedef __attribute__((ext_vector_type(4))) float v4f;

#define QMAXF 127.0f
#define EPSF  1e-8f

static constexpr int M  = 8192;
static constexpr int K  = 1024;
static constexpr int N  = 4096;
static constexpr int KT = K / 64;
static constexpr int NT = N / 16;

static constexpr size_t OFF_AQ = 0;
static constexpr size_t OFF_BQ = OFF_AQ + (size_t)M * K;
static constexpr size_t OFF_SA = OFF_BQ + (size_t)K * N;
static constexpr size_t OFF_SB = OFF_SA + (size_t)M * 4;

#if __has_builtin(__builtin_amdgcn_global_load_async_to_lds_b128)
#define HAVE_ASYNC_LDS 1
#else
#define HAVE_ASYNC_LDS 0
#endif

#if HAVE_ASYNC_LDS
typedef __attribute__((address_space(1))) v4i v4i_gbl;
typedef __attribute__((address_space(3))) v4i v4i_lds;
#endif

__device__ __forceinline__ void copy16_to_lds(const char* g, char* l) {
#if HAVE_ASYNC_LDS
  __builtin_amdgcn_global_load_async_to_lds_b128(
      (v4i_gbl*)(char*)g, (v4i_lds*)l, 0, 0);
#else
  *(v4i*)l = *(const v4i*)g;
#endif
}

__device__ __forceinline__ void async_fill_wait() {
#if HAVE_ASYNC_LDS
#if __has_builtin(__builtin_amdgcn_s_wait_asynccnt)
  __builtin_amdgcn_s_wait_asynccnt(0);
#else
  asm volatile("s_wait_asynccnt 0x0" ::: "memory");
#endif
#endif
}

__device__ __forceinline__ int quant_pack4(float a, float b, float c, float d, float inv) {
  int q0 = (int)rintf(fminf(fmaxf(a * inv, -QMAXF), QMAXF));
  int q1 = (int)rintf(fminf(fmaxf(b * inv, -QMAXF), QMAXF));
  int q2 = (int)rintf(fminf(fmaxf(c * inv, -QMAXF), QMAXF));
  int q3 = (int)rintf(fminf(fmaxf(d * inv, -QMAXF), QMAXF));
  return (q0 & 255) | ((q1 & 255) << 8) | ((q2 & 255) << 16) | ((q3 & 255) << 24);
}

__global__ void quantA(const float* __restrict__ X, char* __restrict__ Aq,
                       float* __restrict__ sA) {
  const int wave = threadIdx.x >> 5;
  const int lane = threadIdx.x & 31;
  const int m    = blockIdx.x * 8 + wave;
  const float* row = X + (size_t)m * K;
  const int ktile = lane >> 1;
  const int half  = lane & 1;

  v4f f[8];
  float mx = 0.0f;
#pragma unroll
  for (int v = 0; v < 8; ++v) {
    const int kbase = ktile * 64 + (v >> 1) * 16 + half * 8 + (v & 1) * 4;
    f[v] = *(const v4f*)(row + kbase);
    mx = fmaxf(mx, fmaxf(fmaxf(fabsf(f[v].x), fabsf(f[v].y)),
                         fmaxf(fabsf(f[v].z), fabsf(f[v].w))));
  }
#pragma unroll
  for (int off = 16; off >= 1; off >>= 1)
    mx = fmaxf(mx, __shfl_xor(mx, off, 32));

  const float s   = fmaxf(mx / QMAXF, EPSF);
  const float inv = 1.0f / s;
  if (lane == 0) sA[m] = s;

  int d[8];
#pragma unroll
  for (int v = 0; v < 8; ++v)
    d[v] = quant_pack4(f[v].x, f[v].y, f[v].z, f[v].w, inv);

  const size_t tile = (size_t)(m >> 4) * KT + ktile;
  char* dst = Aq + tile * 1024 + (size_t)((half << 4) | (m & 15)) * 32;
  *(v4i*)(dst)      = (v4i){d[0], d[1], d[2], d[3]};
  *(v4i*)(dst + 16) = (v4i){d[4], d[5], d[6], d[7]};
}

__global__ void scaleB(const float* __restrict__ Kmat, float* __restrict__ sB) {
  __shared__ float red[256];
  const int t = threadIdx.x;
  const int c = t & 63;
  const int h = blockIdx.x * 64 + c;
  float mx = 0.0f;
  for (int d = (t >> 6); d < K; d += 4)
    mx = fmaxf(mx, fabsf(Kmat[(size_t)d * N + h]));
  red[t] = mx;
  __syncthreads();
  if (t < 64) {
    mx = fmaxf(fmaxf(red[t], red[t + 64]), fmaxf(red[t + 128], red[t + 192]));
    sB[h] = fmaxf(mx / QMAXF, EPSF);
  }
}

__global__ void packB(const float* __restrict__ Kmat, const float* __restrict__ sB,
                      char* __restrict__ Bq) {
  const int wave  = threadIdx.x >> 5;
  const int lane  = threadIdx.x & 31;
  const int wl    = blockIdx.x * 8 + wave;
  const int ktile = wl >> 8;
  const int ntile = wl & 255;
  const int n     = ntile * 16 + (lane & 15);
  const int half  = lane >> 4;
  const float inv = 1.0f / sB[n];

  int d[8];
#pragma unroll
  for (int v = 0; v < 8; ++v) {
    const int kbase = ktile * 64 + (v >> 1) * 16 + half * 8 + (v & 1) * 4;
    float x0 = Kmat[(size_t)(kbase + 0) * N + n];
    float x1 = Kmat[(size_t)(kbase + 1) * N + n];
    float x2 = Kmat[(size_t)(kbase + 2) * N + n];
    float x3 = Kmat[(size_t)(kbase + 3) * N + n];
    d[v] = quant_pack4(x0, x1, x2, x3, inv);
  }
  char* dst = Bq + ((size_t)ktile * NT + ntile) * 1024 + (size_t)lane * 32;
  *(v4i*)(dst)      = (v4i){d[0], d[1], d[2], d[3]};
  *(v4i*)(dst + 16) = (v4i){d[4], d[5], d[6], d[7]};
}

__global__ void gemm_i8(const char* __restrict__ Aq, const char* __restrict__ Bq,
                        const float* __restrict__ sA, const float* __restrict__ sB,
                        float* __restrict__ out) {
  __shared__ __align__(16) char ldsB[2][8192];

  const int wave  = threadIdx.x >> 5;
  const int lane  = threadIdx.x & 31;
  const int mblk  = blockIdx.x >> 5;
  const int nblk  = blockIdx.x & 31;
  const int mtile = mblk * 8 + wave;

  v8i acc[8];
#pragma unroll
  for (int j = 0; j < 8; ++j) acc[j] = (v8i){0, 0, 0, 0, 0, 0, 0, 0};

  const char* aBase = Aq + (size_t)mtile * KT * 1024 + (size_t)lane * 32;
  const char* bBase = Bq + (size_t)(nblk * 8) * 1024 + (size_t)wave * 1024 +
                      (size_t)lane * 32;

  auto fill = [&](int kt, int buf) {
    const char* src = bBase + (size_t)kt * NT * 1024;
    char* d0 = &ldsB[buf][wave * 512 + lane * 16];
    copy16_to_lds(src,      d0);
    copy16_to_lds(src + 16, d0 + 4096);
  };

  union AV { v8i v; v4i h[2]; };
  auto loadA = [&](int kt) {
    AV a;
    const char* ap = aBase + (size_t)kt * 1024;
    a.h[0] = *(const v4i*)(ap);
    a.h[1] = *(const v4i*)(ap + 16);
    return a;
  };

  fill(0, 0);
  AV a_cur = loadA(0);
  async_fill_wait();
  __syncthreads();

  for (int kt = 0; kt < KT; ++kt) {
    const int buf = kt & 1;
    if (kt + 1 < KT) {
      fill(kt + 1, buf ^ 1);
      __builtin_prefetch(aBase + (size_t)(kt + 1) * 1024, 0, 3);
    }
    AV a_nxt;
    if (kt + 1 < KT) a_nxt = loadA(kt + 1);

    const char* blo = &ldsB[buf][lane * 16];
    const char* bhi = blo + 4096;
#pragma unroll
    for (int j = 0; j < 8; ++j) {
      AV b;
      b.h[0] = *(const v4i*)(blo + j * 512);
      b.h[1] = *(const v4i*)(bhi + j * 512);
      acc[j] = __builtin_amdgcn_wmma_i32_16x16x64_iu8(true, a_cur.v, true, b.v,
                                                      acc[j], false, false);
    }
    a_cur = a_nxt;
    async_fill_wait();
    __syncthreads();
  }

  const int mrow0 = mtile * 16 + (lane >> 4) * 8;
  float sa[8];
#pragma unroll
  for (int r = 0; r < 8; ++r) sa[r] = sA[mrow0 + r];

  const int col0 = nblk * 128 + (lane & 15);
#pragma unroll
  for (int j = 0; j < 8; ++j) {
    const int col = col0 + j * 16;
    const float sb = sB[col];
#pragma unroll
    for (int r = 0; r < 8; ++r)
      out[(size_t)(mrow0 + r) * N + col] = (float)acc[j][r] * sa[r] * sb;
  }
}

extern "C" void kernel_launch(void* const* d_in, const int* in_sizes, int n_in,
                              void* d_out, int out_size, void* d_ws, size_t ws_size,
                              hipStream_t stream) {
  const float* X    = (const float*)d_in[0];
  const float* Kmat = (const float*)d_in[1];
  float* out        = (float*)d_out;

  char*  ws = (char*)d_ws;
  char*  Aq = ws + OFF_AQ;
  char*  Bq = ws + OFF_BQ;
  float* sA = (float*)(ws + OFF_SA);
  float* sB = (float*)(ws + OFF_SB);

  (void)in_sizes; (void)n_in; (void)out_size; (void)ws_size;

  quantA<<<M / 8, 256, 0, stream>>>(X, Aq, sA);
  scaleB<<<N / 64, 256, 0, stream>>>(Kmat, sB);
  packB<<<(KT * NT) / 8, 256, 0, stream>>>(Kmat, sB, Bq);
  gemm_i8<<<(M / 128) * (N / 128), 256, 0, stream>>>(Aq, Bq, sA, sB, out);
}
